// BiHMRInputEncoder_22668837388642
// MI455X (gfx1250) — matrix operand order repaired and checked by reading, not compiled, not run on hardware
//
#include <hip/hip_runtime.h>

// ---------------- problem constants ----------------
#define HD     128
#define NGDF   16
#define KNN    16
#define NVERT  40000
#define NGRAPH 8000
#define E1     (NVERT * KNN)    // 640000 vertex-side edges
#define E2     (NGRAPH * KNN)   // 128000 node-side edges
#define BN_EPS 1e-5f

typedef __attribute__((ext_vector_type(16))) __bf16 bf16x16;
typedef __attribute__((ext_vector_type(8)))  __bf16 bf16x8;
typedef __attribute__((ext_vector_type(8)))  float  f32x8;

__device__ __forceinline__ float sigm_(float x)  { return 1.0f / (1.0f + __expf(-x)); }
__device__ __forceinline__ float softp_(float x) { return (x > 20.0f) ? x : log1pf(__expf(x)); }

// ---------------------------------------------------------------------------
// Weight repack: fp32 row-major W[K,N] -> bf16 WMMA-B fragments.
// B (32x16, 16-bit) layout: lane = hf*16 + r holds column n = nt*16 + r,
// element e (of 16) holds K = kt*32 + (e/8)*16 + hf*8 + e%8.  Zero-padded past K.
// ---------------------------------------------------------------------------
__global__ void repack_kernel(const float* __restrict__ W, __bf16* __restrict__ Wp,
                              int Kdim, int Ndim) {
  int nTiles = Ndim >> 4;
  int kTiles = (Kdim + 31) >> 5;
  long long total = (long long)kTiles * nTiles * 512;
  long long i = (long long)blockIdx.x * blockDim.x + threadIdx.x;
  if (i >= total) return;
  int e    = (int)(i & 15);
  int lane = (int)((i >> 4) & 31);
  long long tile = i >> 9;
  int nt = (int)(tile % nTiles);
  int kt = (int)(tile / nTiles);
  int hf = lane >> 4, r = lane & 15;
  int k = kt * 32 + (e >> 3) * 16 + hf * 8 + (e & 7);
  int n = nt * 16 + r;
  float v = (k < Kdim) ? W[(long long)k * Ndim + n] : 0.0f;
  Wp[i] = (__bf16)v;
}

// ---------------------------------------------------------------------------
// A-fragment loader (16x32 bf16 per ISA layout): lane = hf*16 + r holds row m,
// elements 0..7 = K = kb + hf*8 + e, elements 8..15 = K = kb + 16 + hf*8 + e.
// MODE 0: fp32 raw; MODE 1: fp32 + folded BN (sc) + SiLU; MODE 2: bf16 raw.
// Slow path is branchless: clamped address + select-zero.
// ---------------------------------------------------------------------------
template <int MODE>
__device__ __forceinline__ bf16x16 load_a_frag(const void* __restrict__ Aptr, long long m,
                                               int Din, int kb, int hf,
                                               const float* __restrict__ sc, bool fastA) {
  bf16x16 a;
  if (MODE == 2) {
    const __bf16* A = (const __bf16*)Aptr + m * Din;
    if (fastA) {
      bf16x8 c0 = *(const bf16x8*)(A + kb + hf * 8);
      bf16x8 c1 = *(const bf16x8*)(A + kb + 16 + hf * 8);
#pragma unroll
      for (int e = 0; e < 8; ++e) { a[e] = c0[e]; a[8 + e] = c1[e]; }
    } else {
#pragma unroll
      for (int c = 0; c < 2; ++c)
#pragma unroll
        for (int e = 0; e < 8; ++e) {
          int k  = kb + c * 16 + hf * 8 + e;
          int kc = (k < Din) ? k : (Din - 1);        // branchless: always-valid load
          float v = (float)A[kc];
          a[c * 8 + e] = (__bf16)((k < Din) ? v : 0.0f);
        }
    }
  } else {
    const float* A = (const float*)Aptr + m * Din;
    float x[16];
    if (fastA) {
      const float4* q0 = (const float4*)(A + kb + hf * 8);
      const float4* q1 = (const float4*)(A + kb + 16 + hf * 8);
      float4 u0 = q0[0], u1 = q0[1], u2 = q1[0], u3 = q1[1];
      x[0]=u0.x;  x[1]=u0.y;  x[2]=u0.z;  x[3]=u0.w;
      x[4]=u1.x;  x[5]=u1.y;  x[6]=u1.z;  x[7]=u1.w;
      x[8]=u2.x;  x[9]=u2.y;  x[10]=u2.z; x[11]=u2.w;
      x[12]=u3.x; x[13]=u3.y; x[14]=u3.z; x[15]=u3.w;
      if (MODE == 1) {
#pragma unroll
        for (int c = 0; c < 2; ++c)
#pragma unroll
          for (int e = 0; e < 8; ++e) {
            int k = kb + c * 16 + hf * 8 + e;
            float t = sc[k] * x[c * 8 + e] + sc[Din + k];
            x[c * 8 + e] = t * sigm_(t);             // SiLU
          }
      }
    } else {
#pragma unroll
      for (int c = 0; c < 2; ++c)
#pragma unroll
        for (int e = 0; e < 8; ++e) {
          int k  = kb + c * 16 + hf * 8 + e;
          int kc = (k < Din) ? k : (Din - 1);        // branchless
          float v = A[kc];
          if (MODE == 1) {
            float t = sc[kc] * v + sc[Din + kc];
            v = t * sigm_(t);
          }
          x[c * 8 + e] = (k < Din) ? v : 0.0f;
        }
    }
#pragma unroll
    for (int e = 0; e < 16; ++e) a[e] = (__bf16)x[e];
  }
  return a;
}

// ---------------------------------------------------------------------------
// GEMM: Y[N,Dout] = f(A[N,Din]) @ W + bias, via v_wmma_f32_16x16x32_bf16.
// One wave computes a 64x16 output strip (4 M-tiles, one N-tile): the packed-B
// fragment is reused 4x and the 4 WMMAs per k-step are independent chains.
// Block = 8 waves -> 8 N-tiles.
// ---------------------------------------------------------------------------
template <int MODE>
__global__ void __launch_bounds__(256)
gemm_wmma_kernel(const void* __restrict__ Aptr, const __bf16* __restrict__ Wp,
                 const float* __restrict__ bias, const float* __restrict__ sc,
                 float* __restrict__ Y, int N, int Din, int Dout) {
  const int lane = threadIdx.x & 31;
  const int wave = threadIdx.x >> 5;
  const int hf   = lane >> 4;
  const int r    = lane & 15;
  const int nTiles = Dout >> 4;
  const int nTile  = blockIdx.y * 8 + wave;
  if (nTile >= nTiles) return;
  const int  mBase  = blockIdx.x * 64;
  const int  kTiles = (Din + 31) >> 5;
  const bool fastA  = ((Din & 31) == 0);

  long long m[4];
#pragma unroll
  for (int mt = 0; mt < 4; ++mt) {
    int mm = mBase + mt * 16 + r;
    m[mt] = (mm < N) ? mm : (N - 1);    // duplicate a valid row; store is guarded
  }

  f32x8 acc0 = {}, acc1 = {}, acc2 = {}, acc3 = {};

  for (int kt = 0; kt < kTiles; ++kt) {
    const int kb = kt * 32;
    bf16x16 b = *(const bf16x16*)(Wp + (((long long)(kt * nTiles + nTile)) * 32 + lane) * 16);
    bf16x16 a0 = load_a_frag<MODE>(Aptr, m[0], Din, kb, hf, sc, fastA);
    bf16x16 a1 = load_a_frag<MODE>(Aptr, m[1], Din, kb, hf, sc, fastA);
    bf16x16 a2 = load_a_frag<MODE>(Aptr, m[2], Din, kb, hf, sc, fastA);
    bf16x16 a3 = load_a_frag<MODE>(Aptr, m[3], Din, kb, hf, sc, fastA);
    acc0 = __builtin_amdgcn_wmma_f32_16x16x32_bf16(false, a0, false, b, (short)0, acc0, false, false);
    acc1 = __builtin_amdgcn_wmma_f32_16x16x32_bf16(false, a1, false, b, (short)0, acc1, false, false);
    acc2 = __builtin_amdgcn_wmma_f32_16x16x32_bf16(false, a2, false, b, (short)0, acc2, false, false);
    acc3 = __builtin_amdgcn_wmma_f32_16x16x32_bf16(false, a3, false, b, (short)0, acc3, false, false);
  }

  const int col = nTile * 16 + r;
  const float bc = bias[col];
  f32x8 accs[4] = {acc0, acc1, acc2, acc3};
#pragma unroll
  for (int mt = 0; mt < 4; ++mt) {
    const int rowBase = mBase + mt * 16 + hf * 8;   // C/D layout: VGPR i -> M = hf*8 + i
#pragma unroll
    for (int i = 0; i < 8; ++i) {
      int row = rowBase + i;
      if (row < N) Y[(long long)row * Dout + col] = accs[mt][i] + bc;
    }
  }
}

// ---------------------------------------------------------------------------
// Column sum / sum-of-squares over N rows.  blockDim.x == C (128 or 256).
// ---------------------------------------------------------------------------
__global__ void colstats_kernel(const float* __restrict__ Y, float* __restrict__ accum,
                                int N, int C) {
  int col = threadIdx.x;
  int r0 = blockIdx.x * 256;
  int rEnd = r0 + 256; if (rEnd > N) rEnd = N;
  float s = 0.f, ss = 0.f;
  for (int rr = r0; rr < rEnd; ++rr) {
    float v = Y[(long long)rr * C + col];
    s += v; ss += v * v;
  }
  atomicAdd(&accum[col], s);
  atomicAdd(&accum[C + col], ss);
}

// Fold BN into per-column scale/shift: y = s*x + c
__global__ void bnfold_kernel(const float* __restrict__ accum, const float* __restrict__ g,
                              const float* __restrict__ be, float* __restrict__ sc,
                              int C, float invN) {
  int j = blockIdx.x * blockDim.x + threadIdx.x;
  if (j >= C) return;
  float mean = accum[j] * invN;
  float var  = fmaxf(accum[C + j] * invN - mean * mean, 0.0f);
  float s = g[j] * rsqrtf(var + BN_EPS);
  sc[j] = s;
  sc[C + j] = be[j] - mean * s;
}

// gate(BN(Y2)):  Y2[N,256] -> out[N,128]
__global__ void gate_kernel(const float* __restrict__ Y2, const float* __restrict__ sc,
                            float* __restrict__ out, int N) {
  long long i = (long long)blockIdx.x * blockDim.x + threadIdx.x;
  if (i >= (long long)N * HD) return;
  int n = (int)(i >> 7), j = (int)(i & 127);
  float f = sc[j]        * Y2[(long long)n * 256 + j]        + sc[256 + j];
  float c = sc[128 + j]  * Y2[(long long)n * 256 + 128 + j]  + sc[384 + j];
  out[i] = sigm_(f) * softp_(c);
}

// gate(BN(Y2)) summed over 16 contiguous edge rows per segment.
__global__ void gate_segsum_kernel(const float* __restrict__ Y2, const float* __restrict__ sc,
                                   float* __restrict__ out, int Nseg) {
  int j = threadIdx.x;                 // 128 threads
  int v = blockIdx.x;                  // segment
  float sf = sc[j], cf = sc[256 + j], sg = sc[128 + j], cg = sc[384 + j];
  float acc = 0.f;
  for (int i = 0; i < 16; ++i) {
    const float* row = Y2 + ((long long)v * 16 + i) * 256;
    acc += sigm_(sf * row[j] + cf) * softp_(sg * row[128 + j] + cg);
  }
  out[(long long)v * HD + j] = acc;
}

// BN-only output (final MLPs): Y2[N,128] -> out
__global__ void bn_apply_kernel(const float* __restrict__ Y2, const float* __restrict__ sc,
                                float* __restrict__ out, long long total) {
  long long i = (long long)blockIdx.x * blockDim.x + threadIdx.x;
  if (i >= total) return;
  int j = (int)(i & 127);
  out[i] = sc[j] * Y2[i] + sc[128 + j];
}

// cat([A[N,128], B[N,128]]) -> out[N,256]
__global__ void concat_kernel(const float* __restrict__ A, const float* __restrict__ B,
                              float* __restrict__ out, int N) {
  long long i = (long long)blockIdx.x * blockDim.x + threadIdx.x;
  if (i >= (long long)N * 256) return;
  int n = (int)(i >> 8), j = (int)(i & 255);
  out[i] = (j < 128) ? A[(long long)n * 128 + j] : B[(long long)n * 128 + (j - 128)];
}

// ---------------------------------------------------------------------------
// kNN: one thread per query, streaming top-16 (d2) over LDS-tiled refs.
// ---------------------------------------------------------------------------
__global__ void knn_kernel(const float* __restrict__ Q, int NQ,
                           const float* __restrict__ P, int NP, int* __restrict__ outIdx) {
  __shared__ float sx[256], sy[256], sz[256];
  int t = threadIdx.x;
  int q = blockIdx.x * 256 + t;
  bool valid = (q < NQ);
  float qx = 0.f, qy = 0.f, qz = 0.f;
  if (valid) { qx = Q[q * 3]; qy = Q[q * 3 + 1]; qz = Q[q * 3 + 2]; }
  float bd[16]; int bi[16];
#pragma unroll
  for (int s = 0; s < 16; ++s) { bd[s] = 3.4e38f; bi[s] = 0; }

  for (int p0 = 0; p0 < NP; p0 += 256) {
    int p = p0 + t;
    if (p < NP) { sx[t] = P[p * 3]; sy[t] = P[p * 3 + 1]; sz[t] = P[p * 3 + 2]; }
    __syncthreads();
    int lim = NP - p0; if (lim > 256) lim = 256;
    if (valid) {
      for (int j = 0; j < lim; ++j) {
        float dx = qx - sx[j], dy = qy - sy[j], dz = qz - sz[j];
        float d2 = dx * dx + dy * dy + dz * dz;
        if (d2 < bd[15]) {
          float d = d2; int id = p0 + j; bool done = false;
#pragma unroll
          for (int s = 15; s > 0; --s) {
            if (!done) {
              if (d < bd[s - 1]) { bd[s] = bd[s - 1]; bi[s] = bi[s - 1]; }
              else               { bd[s] = d; bi[s] = id; done = true; }
            }
          }
          if (!done) { bd[0] = d; bi[0] = id; }
        }
      }
    }
    __syncthreads();
  }
  if (valid) {
#pragma unroll
    for (int s = 0; s < 16; ++s) outIdx[q * 16 + s] = bi[s];
  }
}

// ---------------------------------------------------------------------------
// Edge features -> bf16 [Ne,160]: [ h[other] (128) | rbf(dist) (16) | rbf(ang) (16) ].
// One wave per edge.  ev = posO[other] - posB[base]; normal from vertex side.
// ---------------------------------------------------------------------------
__global__ void edge_feat_kernel(const float* __restrict__ h, const int* __restrict__ idx,
                                 const float* __restrict__ posO, const float* __restrict__ posB,
                                 const float* __restrict__ normals, int normalFromOther,
                                 __bf16* __restrict__ feat, int Ne) {
  int lane = threadIdx.x & 31;
  long long e = (long long)blockIdx.x * 8 + (threadIdx.x >> 5);
  if (e >= Ne) return;
  int o = idx[e];
  int b = (int)(e >> 4);
  float ex = posO[o * 3]     - posB[b * 3];
  float ey = posO[o * 3 + 1] - posB[b * 3 + 1];
  float ez = posO[o * 3 + 2] - posB[b * 3 + 2];
  float ed = sqrtf(ex * ex + ey * ey + ez * ez);
  int nn = normalFromOther ? o : b;
  float ang = (ex * normals[nn * 3] + ey * normals[nn * 3 + 1] + ez * normals[nn * 3 + 2]) / ed;

  __bf16* fr = feat + e * 160;
#pragma unroll
  for (int t = 0; t < 4; ++t)
    fr[lane + 32 * t] = (__bf16)h[(long long)o * HD + lane + 32 * t];

  float val;
  if (lane < 16) {                               // rbf(ed, 0, 8, 16)
    float mu = (8.0f / 15.0f) * (float)lane;
    float z = (ed - mu) / 0.5f;                  // sigma = 8/16
    val = __expf(-z * z);
  } else {                                       // rbf(ang, -1, 1, 16)
    float mu = -1.0f + (2.0f / 15.0f) * (float)(lane - 16);
    float z = (ang - mu) / 0.125f;               // sigma = 2/16
    val = __expf(-z * z);
  }
  fr[128 + lane] = (__bf16)val;
}

// ===========================================================================
extern "C" void kernel_launch(void* const* d_in, const int* in_sizes, int n_in,
                              void* d_out, int out_size, void* d_ws, size_t ws_size,
                              hipStream_t stream) {
  (void)in_sizes; (void)n_in; (void)out_size; (void)ws_size;

  const float* chem_feats = (const float*)d_in[0];
  const float* geom_feats = (const float*)d_in[1];
  const float* node_pos   = (const float*)d_in[2];
  const float* verts      = (const float*)d_in[3];
  const float* vnormals   = (const float*)d_in[4];

  struct MlpP { const float *W1,*b1,*g1,*be1,*W2,*b2,*g2,*be2; };
  auto getMlp = [&](int base) {
    MlpP p;
    p.W1  = (const float*)d_in[base + 0]; p.b1  = (const float*)d_in[base + 1];
    p.g1  = (const float*)d_in[base + 2]; p.be1 = (const float*)d_in[base + 3];
    p.W2  = (const float*)d_in[base + 4]; p.b2  = (const float*)d_in[base + 5];
    p.g2  = (const float*)d_in[base + 6]; p.be2 = (const float*)d_in[base + 7];
    return p;
  };
  // params dict insertion order: chem, geom, surf_chem, graph_geom, chem_geom, geom_chem
  MlpP Pchem = getMlp(5),  Pgeom  = getMlp(13), Psurf = getMlp(21),
       Pgraph = getMlp(29), Pcg   = getMlp(37), Pgc   = getMlp(45);

  // ---------------- workspace carving ----------------
  char* ws = (char*)d_ws;
  size_t off = 0;
  auto carve = [&](size_t bytes) -> char* {
    char* p = ws + off; off += (bytes + 255) & ~(size_t)255; return p;
  };
  auto pb = [](int K, int N) -> size_t {
    return (size_t)((K + 31) / 32) * (N / 16) * 512 * sizeof(__bf16);
  };

  __bf16* Wp_chem1  = (__bf16*)carve(pb(30, 128));
  __bf16* Wp_chem2  = (__bf16*)carve(pb(128, 256));
  __bf16* Wp_geom1  = (__bf16*)carve(pb(22, 128));
  __bf16* Wp_geom2  = (__bf16*)carve(pb(128, 256));
  __bf16* Wp_surf1  = (__bf16*)carve(pb(160, 128));
  __bf16* Wp_surf2  = (__bf16*)carve(pb(128, 256));
  __bf16* Wp_graph1 = (__bf16*)carve(pb(160, 128));
  __bf16* Wp_graph2 = (__bf16*)carve(pb(128, 256));
  __bf16* Wp_cg1    = (__bf16*)carve(pb(256, 128));
  __bf16* Wp_cg2    = (__bf16*)carve(pb(128, 128));
  __bf16* Wp_gc1    = (__bf16*)carve(pb(256, 128));
  __bf16* Wp_gc2    = (__bf16*)carve(pb(128, 128));

  float* statsAcc = (float*)carve(512 * sizeof(float));
  float* scA      = (float*)carve(512 * sizeof(float));
  float* scB      = (float*)carve(512 * sizeof(float));

  int*   idx_v    = (int*)carve((size_t)E1 * sizeof(int));
  int*   idx_g    = (int*)carve((size_t)E2 * sizeof(int));
  float* h_chem   = (float*)carve((size_t)NGRAPH * HD * sizeof(float));
  float* h_geom   = (float*)carve((size_t)NVERT  * HD * sizeof(float));
  float* h_msg_v  = (float*)carve((size_t)NVERT  * HD * sizeof(float));  // h_chem_geom
  float* h_msg_g  = (float*)carve((size_t)NGRAPH * HD * sizeof(float));  // h_geom_chem

  char* arenaY1  = carve((size_t)E1 * HD * sizeof(float));       // 328 MB
  char* arenaBig = carve((size_t)E1 * 2 * HD * sizeof(float));   // 655 MB

  // ---------------- launch helpers ----------------
  auto repack = [&](const float* W, __bf16* Wp, int K, int N) {
    long long total = (long long)((K + 31) / 32) * (N / 16) * 512;
    repack_kernel<<<dim3((unsigned)((total + 255) / 256)), dim3(256), 0, stream>>>(W, Wp, K, N);
  };
  auto gemm = [&](int mode, const void* A, const __bf16* Wp, const float* bias,
                  const float* sc, float* Y, int N, int Din, int Dout) {
    dim3 grid((N + 63) / 64, ((Dout >> 4) + 7) / 8);
    if (mode == 0)      gemm_wmma_kernel<0><<<grid, 256, 0, stream>>>(A, Wp, bias, sc, Y, N, Din, Dout);
    else if (mode == 1) gemm_wmma_kernel<1><<<grid, 256, 0, stream>>>(A, Wp, bias, sc, Y, N, Din, Dout);
    else                gemm_wmma_kernel<2><<<grid, 256, 0, stream>>>(A, Wp, bias, sc, Y, N, Din, Dout);
  };
  auto bnstats = [&](const float* Y, int N, int C, const float* g, const float* be, float* sc) {
    hipMemsetAsync(statsAcc, 0, (size_t)2 * C * sizeof(float), stream);
    colstats_kernel<<<dim3((N + 255) / 256), dim3(C), 0, stream>>>(Y, statsAcc, N, C);
    bnfold_kernel<<<dim3((C + 127) / 128), dim3(128), 0, stream>>>(statsAcc, g, be, sc, C,
                                                                   1.0f / (float)N);
  };
  auto run_mlp = [&](int modeA, const void* X, int N, int Din, const MlpP& P,
                     const __bf16* W1p, const __bf16* W2p, int dh, int dout,
                     float* Y1, float* Y2) {
    gemm(modeA, X, W1p, P.b1, nullptr, Y1, N, Din, dh);
    bnstats(Y1, N, dh, P.g1, P.be1, scA);
    gemm(1, Y1, W2p, P.b2, scA, Y2, N, dh, dout);     // BN+SiLU fused into A-load
    bnstats(Y2, N, dout, P.g2, P.be2, scB);
  };

  // ---------------- pipeline ----------------
  repack(Pchem.W1,  Wp_chem1, 30, 128);  repack(Pchem.W2,  Wp_chem2, 128, 256);
  repack(Pgeom.W1,  Wp_geom1, 22, 128);  repack(Pgeom.W2,  Wp_geom2, 128, 256);
  repack(Psurf.W1,  Wp_surf1, 160, 128); repack(Psurf.W2,  Wp_surf2, 128, 256);
  repack(Pgraph.W1, Wp_graph1,160, 128); repack(Pgraph.W2, Wp_graph2,128, 256);
  repack(Pcg.W1,    Wp_cg1,   256, 128); repack(Pcg.W2,    Wp_cg2,   128, 128);
  repack(Pgc.W1,    Wp_gc1,   256, 128); repack(Pgc.W2,    Wp_gc2,   128, 128);

  float* Y1 = (float*)arenaY1;
  float* Y2 = (float*)arenaBig;

  // h_chem = gate(mlp(chem_feats))
  run_mlp(0, chem_feats, NGRAPH, 30, Pchem, Wp_chem1, Wp_chem2, 128, 256, Y1, Y2);
  gate_kernel<<<dim3((unsigned)(((long long)NGRAPH * HD + 255) / 256)), 256, 0, stream>>>(
      Y2, scB, h_chem, NGRAPH);

  // h_geom = gate(mlp(geom_feats))
  run_mlp(0, geom_feats, NVERT, 22, Pgeom, Wp_geom1, Wp_geom2, 128, 256, Y1, Y2);
  gate_kernel<<<dim3((unsigned)(((long long)NVERT * HD + 255) / 256)), 256, 0, stream>>>(
      Y2, scB, h_geom, NVERT);

  // kNN both directions
  knn_kernel<<<dim3((NVERT + 255) / 256), 256, 0, stream>>>(verts, NVERT, node_pos, NGRAPH, idx_v);
  knn_kernel<<<dim3((NGRAPH + 255) / 256), 256, 0, stream>>>(node_pos, NGRAPH, verts, NVERT, idx_g);

  // chem -> surface messages
  __bf16* feat1 = (__bf16*)arenaBig;
  edge_feat_kernel<<<dim3((E1 + 7) / 8), 256, 0, stream>>>(
      h_chem, idx_v, node_pos, verts, vnormals, /*normalFromOther=*/0, feat1, E1);
  run_mlp(2, feat1, E1, 160, Psurf, Wp_surf1, Wp_surf2, 128, 256, Y1, Y2);  // Y2 overwrites feat1 (dead)
  gate_segsum_kernel<<<dim3(NVERT), 128, 0, stream>>>(Y2, scB, h_msg_v, NVERT);

  // h_geom_mix = mlp(cat([h_chem_geom, h_geom]))  -> d_out[0 .. NV*H)
  float* cat1 = (float*)arenaY1;
  concat_kernel<<<dim3((unsigned)(((long long)NVERT * 256 + 255) / 256)), 256, 0, stream>>>(
      h_msg_v, h_geom, cat1, NVERT);
  float* Y1b = (float*)arenaBig;
  float* Y2b = (float*)(arenaBig + ((size_t)64 << 20));
  run_mlp(0, cat1, NVERT, 256, Pcg, Wp_cg1, Wp_cg2, 128, 128, Y1b, Y2b);
  bn_apply_kernel<<<dim3((unsigned)(((long long)NVERT * HD + 255) / 256)), 256, 0, stream>>>(
      Y2b, scB, (float*)d_out, (long long)NVERT * HD);

  // geom -> graph messages
  __bf16* feat2 = (__bf16*)arenaBig;
  edge_feat_kernel<<<dim3((E2 + 7) / 8), 256, 0, stream>>>(
      h_geom, idx_g, verts, node_pos, vnormals, /*normalFromOther=*/1, feat2, E2);
  float* Y2g = (float*)(arenaBig + ((size_t)256 << 20));
  run_mlp(2, feat2, E2, 160, Pgraph, Wp_graph1, Wp_graph2, 128, 256, Y1, Y2g);
  gate_segsum_kernel<<<dim3(NGRAPH), 128, 0, stream>>>(Y2g, scB, h_msg_g, NGRAPH);

  // h_chem_mix = mlp(cat([h_geom_chem, h_chem]))  -> d_out[NV*H ..)
  float* cat2 = (float*)arenaY1;
  concat_kernel<<<dim3((unsigned)(((long long)NGRAPH * 256 + 255) / 256)), 256, 0, stream>>>(
      h_msg_g, h_chem, cat2, NGRAPH);
  run_mlp(0, cat2, NGRAPH, 256, Pgc, Wp_gc1, Wp_gc2, 128, 128, Y1b, Y2b);
  bn_apply_kernel<<<dim3((unsigned)(((long long)NGRAPH * HD + 255) / 256)), 256, 0, stream>>>(
      Y2b, scB, (float*)d_out + (size_t)NVERT * HD, (long long)NGRAPH * HD);
}
